// Block_42709154791497
// MI455X (gfx1250) — matrix operand order repaired and checked by reading, not compiled, not run on hardware
//
#include <hip/hip_runtime.h>
#include <hip/hip_bf16.h>

typedef __bf16 bf16;
typedef __attribute__((ext_vector_type(16))) __bf16 v16bf;
typedef __attribute__((ext_vector_type(8)))  float  v8f;
typedef __attribute__((ext_vector_type(4)))  unsigned int v4u;
typedef __attribute__((ext_vector_type(8)))  int v8i;
typedef __attribute__((ext_vector_type(4)))  int v4i;

#define DIM      1024
#define HEADS    16
#define HEAD_DIM 64
#define HIDDEN   4096
#define BATCH    8
#define SEQ      1024
#define TOKENS   (BATCH*SEQ)

struct __align__(16) U4 { unsigned int a, b, c, d; };
union Frag { v16bf v; U4 q[2]; bf16 e[16]; };

__device__ __forceinline__ v8f wmma_bf16(v16bf a, v16bf b, v8f c) {
  return __builtin_amdgcn_wmma_f32_16x16x32_bf16(
      /*neg_a=*/false, a, /*neg_b=*/false, b,
      /*c_mod=*/(short)0, c, /*reuse_a=*/false, /*reuse_b=*/false);
}

// A-fragment: 16x32 bf16, row-major k-contiguous source.
// lane<16: row=lane, K {k0+0..7, k0+16..23}; lane>=16: same row, K {k0+8..15, k0+24..31}
__device__ __forceinline__ v16bf load_frag_a(const bf16* rowk0, int half) {
  Frag f;
  f.q[0] = *(const U4*)(rowk0 + half * 8);
  f.q[1] = *(const U4*)(rowk0 + 16 + half * 8);
  return f.v;
}
// B-fragment: 32x16 bf16 from k-contiguous-per-column source.
// lane<16: col=lane, K {k0+0..7, k0+16..23}; lane>=16: col=lane-16, K {k0+8..15, k0+24..31}
__device__ __forceinline__ v16bf load_frag_b(const bf16* colk0, int half) {
  Frag f;
  f.q[0] = *(const U4*)(colk0 + half * 8);
  f.q[1] = *(const U4*)(colk0 + 16 + half * 8);
  return f.v;
}

// ---------------------------------------------------------------------------
// TDM: DMA a 2D bf16 tile (tile_k x tile_rows, row stride = stride_elems)
// from global memory into LDS, padding each 64-byte row with 32 bytes so the
// LDS row stride is 48 elements (96B) for bank-conflict-friendly fragment
// reads. D# packing per CDNA5 ISA.
// ---------------------------------------------------------------------------
__device__ __forceinline__ void tdm_load_tile_2d(
    unsigned lds_off, const bf16* gptr,
    unsigned tile_k, unsigned tile_rows, unsigned stride_elems) {
  unsigned long long ga = (unsigned long long)(const void*)gptr;
  v4u g0;
  g0[0] = 1u;                                            // count=1, user mode
  g0[1] = lds_off;                                       // lds_addr (bytes)
  g0[2] = (unsigned)ga;                                  // global_addr[31:0]
  g0[3] = (unsigned)((ga >> 32) & 0x1FFFFFFu) | (2u << 30);  // addr[56:32], type=2
  const unsigned td0 = 0x40000000u, td1 = 0x40000000u;   // huge dims: no OOB clip
  v8i g1;
  g1[0] = (int)((1u << 16)      // data_size = 1 -> 2 bytes
              | (1u << 20)      // pad_enable
              | (3u << 22)      // pad_interval: after 16 DWORDs (one 64B row)
              | (7u << 25));    // pad_amount: 8 DWORDs (32B) -> 96B row stride
  g1[1] = (int)((td0 & 0xFFFFu) << 16);                  // tensor_dim0[15:0]
  g1[2] = (int)((td0 >> 16) | ((td1 & 0xFFFFu) << 16));  // td0 hi | td1 lo
  g1[3] = (int)((td1 >> 16) | (tile_k << 16));           // td1 hi | tile_dim0
  g1[4] = (int)tile_rows;                                // tile_dim1 (tile_dim2=0)
  g1[5] = (int)stride_elems;                             // tensor_dim0_stride lo
  g1[6] = 0;
  g1[7] = 0;
  v4i z4 = {0, 0, 0, 0};
  v8i z8 = {0, 0, 0, 0, 0, 0, 0, 0};
  __builtin_amdgcn_tensor_load_to_lds(g0, g1, z4, z4, z8, 0);
}

__device__ __forceinline__ unsigned lds_offset_of(const void* p) {
  return (unsigned)(size_t)p;  // LDS aperture: addr[31:0] is the LDS offset
}

__device__ __forceinline__ float gelu_tanh(float x) {
  float x3 = x * x * x;
  return 0.5f * x * (1.0f + tanhf(0.7978845608028654f * (x + 0.044715f * x3)));
}

#define EPI_BF16_BIAS      0
#define EPI_BF16_BIAS_GELU 1
#define EPI_F32_BIAS_RES   2

// ---------------------------------------------------------------------------
// bf16 WMMA GEMM with TDM-staged, double-buffered LDS tiles.
// C = A[MxK] * Bt[NxK]^T  (Bt pre-transposed, k-contiguous rows).
// Block tile 128x64, 8 waves (4x2), each wave 32x32 via 2x2 WMMA accumulators.
// Wave 0 drives the Tensor Data Mover; everyone computes.
// ---------------------------------------------------------------------------
template <int EPI>
__global__ __launch_bounds__(256, 2) void gemm_bf16_kernel(
    const bf16* __restrict__ A, const bf16* __restrict__ Bt,
    const float* __restrict__ bias, const float* __restrict__ res,
    void* __restrict__ outp, int M, int N, int K) {
  __shared__ __align__(16) bf16 As[2][128 * 48];  // row stride 48 via TDM pad
  __shared__ __align__(16) bf16 Bs[2][64 * 48];

  const int tid  = threadIdx.x;
  const int lane = tid & 31, wave = tid >> 5;
  const int half = lane >> 4, l16 = lane & 15;
  const int mblk = blockIdx.y * 128, nblk = blockIdx.x * 64;
  const int mw = (wave >> 1) * 32, nw = (wave & 1) * 32;

  const bf16* Ag = A  + (size_t)mblk * K;
  const bf16* Bg = Bt + (size_t)nblk * K;

  v8f acc[2][2] = {{{}, {}}, {{}, {}}};
  const int ntiles = K >> 5;

  if (wave == 0) {
    tdm_load_tile_2d(lds_offset_of(&As[0][0]), Ag, 32, 128, (unsigned)K);
    tdm_load_tile_2d(lds_offset_of(&Bs[0][0]), Bg, 32, 64,  (unsigned)K);
    __builtin_amdgcn_s_wait_tensorcnt(0);
  }
  __syncthreads();

  for (int t = 0; t < ntiles; t++) {
    const int cur = t & 1, nxt = cur ^ 1;
    if (wave == 0 && t + 1 < ntiles) {   // prefetch next K-tile via TDM
      int k0n = (t + 1) << 5;
      tdm_load_tile_2d(lds_offset_of(&As[nxt][0]), Ag + k0n, 32, 128, (unsigned)K);
      tdm_load_tile_2d(lds_offset_of(&Bs[nxt][0]), Bg + k0n, 32, 64,  (unsigned)K);
    }

    v16bf a0 = load_frag_a(&As[cur][(mw + l16) * 48], half);
    v16bf a1 = load_frag_a(&As[cur][(mw + 16 + l16) * 48], half);
    v16bf b0 = load_frag_b(&Bs[cur][(nw + l16) * 48], half);
    v16bf b1 = load_frag_b(&Bs[cur][(nw + 16 + l16) * 48], half);
    acc[0][0] = wmma_bf16(a0, b0, acc[0][0]);
    acc[0][1] = wmma_bf16(a0, b1, acc[0][1]);
    acc[1][0] = wmma_bf16(a1, b0, acc[1][0]);
    acc[1][1] = wmma_bf16(a1, b1, acc[1][1]);

    if (wave == 0) __builtin_amdgcn_s_wait_tensorcnt(0);
    __syncthreads();
  }

#pragma unroll
  for (int mi = 0; mi < 2; mi++) {
#pragma unroll
    for (int ni = 0; ni < 2; ni++) {
      int col = nblk + nw + ni * 16 + l16;
      float bb = bias[col];
#pragma unroll
      for (int r = 0; r < 8; r++) {
        int row = mblk + mw + mi * 16 + half * 8 + r;
        size_t idx = (size_t)row * N + col;
        float v = acc[mi][ni][r] + bb;
        if (EPI == EPI_F32_BIAS_RES)        ((float*)outp)[idx] = v + res[idx];
        else if (EPI == EPI_BF16_BIAS_GELU) ((bf16*)outp)[idx] = (bf16)gelu_tanh(v);
        else                                ((bf16*)outp)[idx] = (bf16)v;
      }
    }
  }
}

// ---------------------------------------------------------------------------
// Flash attention: one block = 64 q-rows of one (batch, head); 4 waves,
// each wave owns 16 q-rows. Online softmax, WMMA for QK^T and PV.
// qkv layout: [token][3*DIM] bf16, q at h*64, k at DIM+h*64, v at 2*DIM+h*64.
// ---------------------------------------------------------------------------
__global__ __launch_bounds__(128, 2) void attention_kernel(
    const bf16* __restrict__ qkv, bf16* __restrict__ optr) {
  __shared__ __align__(16) bf16 Vt[64 * 72];        // [d][j] transposed V tile
  __shared__ __align__(16) bf16 Pb[4 * 16 * 72];    // per-wave P tiles

  const int tid  = threadIdx.x;
  const int lane = tid & 31, wave = tid >> 5;
  const int half = lane >> 4, l16 = lane & 15;
  const int qt = blockIdx.x, h = blockIdx.y, b = blockIdx.z;
  const int bq = b * SEQ;
  const int q0 = qt * 64 + wave * 16;
  bf16* Pw = Pb + wave * 16 * 72;

  const bf16* qrow = qkv + (size_t)(bq + q0 + l16) * (3 * DIM) + h * HEAD_DIM;
  v16bf aq0 = load_frag_a(qrow, half);       // d 0..31
  v16bf aq1 = load_frag_a(qrow + 32, half);  // d 32..63

  v8f o[4] = {{}, {}, {}, {}};
  float m[8], l[8];
#pragma unroll
  for (int r = 0; r < 8; r++) { m[r] = -1e30f; l[r] = 0.0f; }

  for (int jt = 0; jt < SEQ / 64; jt++) {
    __syncthreads();  // previous iteration's Vt reads done before overwrite
#pragma unroll
    for (int i = 0; i < 4; i++) {
      int cc = tid + i * 128;                // 512 chunks of 8 elems
      int jj = cc >> 3, dc = (cc & 7) * 8;
      Frag vf;
      vf.q[0] = *(const U4*)(qkv + (size_t)(bq + jt * 64 + jj) * (3 * DIM) +
                             2 * DIM + h * HEAD_DIM + dc);
#pragma unroll
      for (int e = 0; e < 8; e++) Vt[(dc + e) * 72 + jj] = vf.e[e];
    }
    __syncthreads();

    // S = Q K^T  (K rows are d-contiguous in global -> direct B fragments)
    v8f s[4] = {{}, {}, {}, {}};
#pragma unroll
    for (int c = 0; c < 4; c++) {
      int j = jt * 64 + c * 16 + l16;
      const bf16* krow = qkv + (size_t)(bq + j) * (3 * DIM) + DIM + h * HEAD_DIM;
      v16bf bk0 = load_frag_b(krow, half);
      v16bf bk1 = load_frag_b(krow + 32, half);
      s[c] = wmma_bf16(aq0, bk0, s[c]);
      s[c] = wmma_bf16(aq1, bk1, s[c]);
    }
    const float scale = 0.125f;  // HEAD_DIM^-0.5
#pragma unroll
    for (int c = 0; c < 4; c++)
#pragma unroll
      for (int r = 0; r < 8; r++) s[c][r] *= scale;

    // online softmax; each row spans the 16 lanes of a half-wave
#pragma unroll
    for (int r = 0; r < 8; r++) {
      float a = fmaxf(fmaxf(s[0][r], s[1][r]), fmaxf(s[2][r], s[3][r]));
#pragma unroll
      for (int off = 1; off < 16; off <<= 1) a = fmaxf(a, __shfl_xor(a, off, 16));
      float mn = fmaxf(m[r], a);
      float alpha = __expf(m[r] - mn);
      m[r] = mn;
      float rs = 0.0f;
#pragma unroll
      for (int c = 0; c < 4; c++) {
        float p = __expf(s[c][r] - mn);
        s[c][r] = p;
        rs += p;
      }
#pragma unroll
      for (int off = 1; off < 16; off <<= 1) rs += __shfl_xor(rs, off, 16);
      l[r] = l[r] * alpha + rs;
#pragma unroll
      for (int c = 0; c < 4; c++) o[c][r] *= alpha;
    }

    // P (C-layout) -> LDS -> re-read as A fragments
#pragma unroll
    for (int c = 0; c < 4; c++)
#pragma unroll
      for (int r = 0; r < 8; r++)
        Pw[(half * 8 + r) * 72 + c * 16 + l16] = (bf16)s[c][r];
    __syncthreads();

    const bf16* prow = Pw + l16 * 72;
    v16bf ap0 = load_frag_a(prow, half);       // j 0..31
    v16bf ap1 = load_frag_a(prow + 32, half);  // j 32..63
#pragma unroll
    for (int c = 0; c < 4; c++) {
      const bf16* vcol = Vt + (c * 16 + l16) * 72;
      v16bf bv0 = load_frag_b(vcol, half);
      v16bf bv1 = load_frag_b(vcol + 32, half);
      o[c] = wmma_bf16(ap0, bv0, o[c]);
      o[c] = wmma_bf16(ap1, bv1, o[c]);
    }
  }

#pragma unroll
  for (int c = 0; c < 4; c++) {
    int col = h * HEAD_DIM + c * 16 + l16;
#pragma unroll
    for (int r = 0; r < 8; r++) {
      int row = q0 + half * 8 + r;
      optr[(size_t)(bq + row) * DIM + col] = (bf16)(o[c][r] / l[r]);
    }
  }
}

// ---------------------------------------------------------------------------
// LayerNorm over last dim (1024), fp32 in -> bf16 out. One block per row.
// ---------------------------------------------------------------------------
__global__ __launch_bounds__(256) void layernorm_bf16_kernel(
    const float* __restrict__ x, const float* __restrict__ scale,
    const float* __restrict__ bias, bf16* __restrict__ out) {
  __shared__ float red[256];
  const int row = blockIdx.x;
  const size_t base = (size_t)row * DIM;
  const int tid = threadIdx.x;
  float v[4];
  float s = 0.0f;
#pragma unroll
  for (int i = 0; i < 4; i++) { v[i] = x[base + tid + i * 256]; s += v[i]; }
  red[tid] = s;
  __syncthreads();
  for (int st = 128; st > 0; st >>= 1) {
    if (tid < st) red[tid] += red[tid + st];
    __syncthreads();
  }
  float mu = red[0] * (1.0f / DIM);
  __syncthreads();
  float ss = 0.0f;
#pragma unroll
  for (int i = 0; i < 4; i++) { float d = v[i] - mu; ss += d * d; }
  red[tid] = ss;
  __syncthreads();
  for (int st = 128; st > 0; st >>= 1) {
    if (tid < st) red[tid] += red[tid + st];
    __syncthreads();
  }
  float inv = rsqrtf(red[0] * (1.0f / DIM) + 1e-6f);
#pragma unroll
  for (int i = 0; i < 4; i++) {
    int col = tid + i * 256;
    out[base + col] = (bf16)((v[i] - mu) * inv * scale[col] + bias[col]);
  }
}

// ---------------------------------------------------------------------------
// One-time weight prep: src[K][N] fp32 -> dst[N][K] bf16 (LDS-tiled transpose)
// so every GEMM B-tile is a k-contiguous row block, directly TDM-loadable.
// ---------------------------------------------------------------------------
__global__ __launch_bounds__(256) void transpose_f32_to_bf16_kernel(
    const float* __restrict__ src, bf16* __restrict__ dst, int K, int N) {
  __shared__ float tile[32][33];
  const int kb = blockIdx.y * 32, nb = blockIdx.x * 32;
  const int tx = threadIdx.x & 31, ty = threadIdx.x >> 5;  // 32 x 8
#pragma unroll
  for (int i = 0; i < 32; i += 8)
    tile[ty + i][tx] = src[(size_t)(kb + ty + i) * N + nb + tx];
  __syncthreads();
#pragma unroll
  for (int i = 0; i < 32; i += 8)
    dst[(size_t)(nb + ty + i) * K + kb + tx] = (bf16)tile[tx][ty + i];
}

// ---------------------------------------------------------------------------
extern "C" void kernel_launch(void* const* d_in, const int* in_sizes, int n_in,
                              void* d_out, int out_size, void* d_ws,
                              size_t ws_size, hipStream_t stream) {
  const float* x      = (const float*)d_in[0];
  const float* w_qkv  = (const float*)d_in[1];
  const float* b_qkv  = (const float*)d_in[2];
  const float* w_proj = (const float*)d_in[3];
  const float* b_proj = (const float*)d_in[4];
  const float* ln1_s  = (const float*)d_in[5];
  const float* ln1_b  = (const float*)d_in[6];
  const float* ln2_s  = (const float*)d_in[7];
  const float* ln2_b  = (const float*)d_in[8];
  const float* w_fc1  = (const float*)d_in[9];
  const float* b_fc1  = (const float*)d_in[10];
  const float* w_fc2  = (const float*)d_in[11];
  const float* b_fc2  = (const float*)d_in[12];

  char* ws = (char*)d_ws;
  size_t off = 0;
  auto alloc = [&](size_t bytes) {
    char* p = ws + off;
    off += (bytes + 255) & ~(size_t)255;
    return p;
  };
  bf16* wqkvT  = (bf16*)alloc((size_t)3 * DIM * DIM * 2);  // [3C][C]
  bf16* wprojT = (bf16*)alloc((size_t)DIM * DIM * 2);      // [C][C]
  bf16* wfc1T  = (bf16*)alloc((size_t)HIDDEN * DIM * 2);   // [H][C]
  bf16* wfc2T  = (bf16*)alloc((size_t)DIM * HIDDEN * 2);   // [C][H]
  bf16* h_bf   = (bf16*)alloc((size_t)TOKENS * DIM * 2);
  bf16* qkv_bf = (bf16*)alloc((size_t)TOKENS * 3 * DIM * 2);
  bf16* o_bf   = (bf16*)alloc((size_t)TOKENS * DIM * 2);
  float* x1    = (float*)alloc((size_t)TOKENS * DIM * 4);
  bf16* h2_bf  = (bf16*)alloc((size_t)TOKENS * DIM * 2);
  bf16* hid_bf = (bf16*)alloc((size_t)TOKENS * HIDDEN * 2);

  // one-time weight transposes (fp32 [K][N] -> bf16 [N][K])
  transpose_f32_to_bf16_kernel<<<dim3(3 * DIM / 32, DIM / 32), 256, 0, stream>>>(
      w_qkv, wqkvT, DIM, 3 * DIM);
  transpose_f32_to_bf16_kernel<<<dim3(DIM / 32, DIM / 32), 256, 0, stream>>>(
      w_proj, wprojT, DIM, DIM);
  transpose_f32_to_bf16_kernel<<<dim3(HIDDEN / 32, DIM / 32), 256, 0, stream>>>(
      w_fc1, wfc1T, DIM, HIDDEN);
  transpose_f32_to_bf16_kernel<<<dim3(DIM / 32, HIDDEN / 32), 256, 0, stream>>>(
      w_fc2, wfc2T, HIDDEN, DIM);

  // attention branch
  layernorm_bf16_kernel<<<TOKENS, 256, 0, stream>>>(x, ln1_s, ln1_b, h_bf);
  gemm_bf16_kernel<EPI_BF16_BIAS><<<dim3(3 * DIM / 64, TOKENS / 128), 256, 0, stream>>>(
      h_bf, wqkvT, b_qkv, nullptr, qkv_bf, TOKENS, 3 * DIM, DIM);
  attention_kernel<<<dim3(SEQ / 64, HEADS, BATCH), 128, 0, stream>>>(qkv_bf, o_bf);
  gemm_bf16_kernel<EPI_F32_BIAS_RES><<<dim3(DIM / 64, TOKENS / 128), 256, 0, stream>>>(
      o_bf, wprojT, b_proj, x, x1, TOKENS, DIM, DIM);

  // mlp branch
  layernorm_bf16_kernel<<<TOKENS, 256, 0, stream>>>(x1, ln2_s, ln2_b, h2_bf);
  gemm_bf16_kernel<EPI_BF16_BIAS_GELU><<<dim3(HIDDEN / 64, TOKENS / 128), 256, 0, stream>>>(
      h2_bf, wfc1T, b_fc1, nullptr, hid_bf, TOKENS, HIDDEN, DIM);
  gemm_bf16_kernel<EPI_F32_BIAS_RES><<<dim3(DIM / 64, TOKENS / 128), 256, 0, stream>>>(
      hid_bf, wfc2T, b_fc2, x1, (float*)d_out, TOKENS, DIM, HIDDEN);
}
